// QuantizationLayerEST_824633721183
// MI455X (gfx1250) — compile-verified, matrix operand order checked by reading, not run on hardware
//
#include <hip/hip_runtime.h>

typedef __attribute__((ext_vector_type(16))) _Float16 v16h;
typedef __attribute__((ext_vector_type(2)))  _Float16 h2v;
typedef __attribute__((ext_vector_type(8)))  float    v8f;

#define CC 9
#define HH 180
#define WW 240
#define HIDN 100
#define SLOPEF 0.1f

// workspace byte offsets
#define OFF_TMAX 0       // 16 x u32 (float bits)
#define OFF_W3   64      // 128 x f32 (zero padded)
#define OFF_B3   576     // 1 x f32
#define OFF_W1PK 640     // 64 x f16-pair (fragment-ordered w1 pairs)
#define OFF_B1PK 896     // 64 x f16-pair (fragment-ordered b1 pairs, b1[100]:=1.0 for b2-fold)
#define OFF_FRAG 1152    // 28 frags * 32 lanes * 16 f16 = 28672 bytes (w2^T + b2 row, A layout)

// ---------------- prep: tables + w2^T/b2 -> WMMA-A f16 fragment layout ----------------
__global__ void prep_kernel(const float* __restrict__ w1, const float* __restrict__ b1,
                            const float* __restrict__ w2, const float* __restrict__ b2,
                            const float* __restrict__ w3, const float* __restrict__ b3,
                            unsigned char* __restrict__ ws) {
  int tid = blockIdx.x * blockDim.x + threadIdx.x;
  int nth = gridDim.x * blockDim.x;
  unsigned* tmax = (unsigned*)(ws + OFF_TMAX);
  float* tw3 = (float*)(ws + OFF_W3);
  if (tid < 16) tmax[tid] = 0u;
  if (tid == 0) ((float*)(ws + OFF_B3))[0] = b3[0];
  for (int k = tid; k < 128; k += nth) tw3[k] = (k < HIDN) ? w3[k] : 0.f;

  // fragment-ordered (w1, b1) f16 pair tables; b1 entry at k==100 is 1.0 (bias-fold row)
  h2v* pw1 = (h2v*)(ws + OFF_W1PK);
  h2v* pb1 = (h2v*)(ws + OFF_B1PK);
  for (int idx = tid; idx < 64; idx += nth) {
    int half = idx >> 5, kb = (idx >> 3) & 3, i = idx & 7;
    int k0 = 32 * kb + ((i & 4) << 2) + half * 8 + ((2 * i) & 7);
    float wa = (k0     < HIDN) ? w1[k0]     : 0.f;
    float wb = (k0 + 1 < HIDN) ? w1[k0 + 1] : 0.f;
    float ba = (k0     == HIDN) ? 1.f : ((k0     < HIDN) ? b1[k0]     : 0.f);
    float bb = (k0 + 1 == HIDN) ? 1.f : ((k0 + 1 < HIDN) ? b1[k0 + 1] : 0.f);
    h2v wp = {(_Float16)wa, (_Float16)wb};
    h2v bp = {(_Float16)ba, (_Float16)bb};
    pw1[idx] = wp;
    pb1[idx] = bp;
  }

  // w2^T tiles in WMMA A layout: lane -> row j, element -> k; k==100 row holds b2[j]
  _Float16* frag = (_Float16*)(ws + OFF_FRAG);
  for (int g = tid; g < 28 * 32 * 16; g += nth) {
    int e    = g & 15;
    int lane = (g >> 4) & 31;
    int f    = g >> 9;           // 0..27
    int jt   = f >> 2, kb = f & 3;
    int half = lane >> 4;
    int j    = (lane & 15) + 16 * jt;                         // output neuron (row)
    int k    = 32 * kb + ((e & 8) << 1) + half * 8 + (e & 7); // input neuron (col of w2^T)
    float v = 0.f;
    if (j < HIDN) {
      if (k < HIDN)       v = w2[k * HIDN + j];   // w2^T[j][k]
      else if (k == HIDN) v = b2[j];              // folded bias row
    }
    frag[g] = (_Float16)v;
  }
}

// ---------------- per-bin segment max of t (b is sorted) ----------------
__global__ void binmax_kernel(const float* __restrict__ t, const int* __restrict__ b,
                              int n, unsigned* __restrict__ tmaxb) {
  int i  = blockIdx.x * blockDim.x + threadIdx.x;
  int ic = (i < n) ? i : (n - 1);
  unsigned bits = (i < n) ? __float_as_uint(t[ic]) : 0u;  // t >= 0 -> bit-monotonic
  int bi = b[ic];
  int b0  = __shfl(bi, 0);
  int b31 = __shfl(bi, 31);
  if (b0 == b31) {
    #pragma unroll
    for (int mask = 16; mask > 0; mask >>= 1) {
      unsigned o = __shfl_xor(bits, mask);
      bits = (bits > o) ? bits : o;
    }
    if ((threadIdx.x & 31) == 0) atomicMax(tmaxb + bi, bits);
  } else {
    atomicMax(tmaxb + bi, bits);
  }
}

// ---------------- main: transposed GEMM, 16-event tiles, 9 channels ----------------
__global__ __launch_bounds__(256) void evmlp_kernel(
    const float* __restrict__ tp, const int* __restrict__ xp,
    const int* __restrict__ yp,  const int* __restrict__ pp,
    const int* __restrict__ bp,  const unsigned char* __restrict__ ws,
    float* __restrict__ out, int n) {
  const int tid   = threadIdx.x;
  const int lane  = tid & 31;
  const int wid   = tid >> 5;
  const int m     = lane & 15;
  const int half8 = (lane >> 4) << 3;

  const float b3v = ((const float*)(ws + OFF_B3))[0];
  const unsigned* tmaxb = (const unsigned*)(ws + OFF_TMAX);

  // hoist per-lane layer-1 weight pairs (f16) into registers
  h2v w1r[32], b1r[32];
  {
    const h2v* pw1 = (const h2v*)(ws + OFF_W1PK) + (lane >> 4) * 32;
    const h2v* pb1 = (const h2v*)(ws + OFF_B1PK) + (lane >> 4) * 32;
    #pragma unroll
    for (int q = 0; q < 32; ++q) { w1r[q] = pw1[q]; b1r[q] = pb1[q]; }
  }
  // hoist per-lane w3 values: j = jt*16 + r + half8
  float w3r[56];
  {
    const float* w3t = (const float*)(ws + OFF_W3);
    #pragma unroll
    for (int jt = 0; jt < 7; ++jt)
      #pragma unroll
      for (int r = 0; r < 8; ++r)
        w3r[jt * 8 + r] = w3t[jt * 16 + r + half8];
  }
  // w2^T (+b2 row) as WMMA A operand, resident in VGPRs: 28 x v16h
  v16h W2A[28];
  {
    const unsigned char* fb = ws + OFF_FRAG;
    #pragma unroll
    for (int f = 0; f < 28; ++f) {
      union { uint4 q[2]; v16h v; } u;
      const uint4* p = (const uint4*)(fb + (size_t)(f * 32 + lane) * 32u);
      u.q[0] = p[0];
      u.q[1] = p[1];
      W2A[f] = u.v;
    }
  }

  const h2v c01 = {(_Float16)SLOPEF, (_Float16)SLOPEF};
  const int ntiles  = (n + 15) >> 4;
  const int wstride = gridDim.x * 8;
  for (int tile = blockIdx.x * 8 + wid; tile < ntiles; tile += wstride) {
    int ev = tile * 16 + m;
    bool valid = (ev < n);
    int ec = valid ? ev : (n - 1);
    float tv = tp[ec];
    int  bi  = bp[ec];
    float tmax = __uint_as_float(tmaxb[bi]);
    float tn = tv / tmax;
    int xi = xp[ec], yi = yp[ec], pi = pp[ec];
    int p01  = (pi + 1) >> 1;
    int idx0 = xi + WW * yi + WW * HH * CC * p01 + WW * HH * CC * 2 * bi;

    #pragma unroll 1
    for (int c = 0; c < CC; ++c) {
      float s = tn - (float)c * 0.125f;
      _Float16 sh = (_Float16)s;
      h2v s2 = {sh, sh};

      // layer 1 in packed f16, built straight into the WMMA B layout
      // (lane -> event column m, elements -> k; k==100 slot yields 1.0 for b2-fold)
      v16h Hb[4];
      #pragma unroll
      for (int kb = 0; kb < 4; ++kb) {
        union { h2v h2[8]; v16h v; } bv;
        #pragma unroll
        for (int i = 0; i < 8; ++i) {
          h2v h = __builtin_elementwise_fma(s2, w1r[kb * 8 + i], b1r[kb * 8 + i]);
          h = __builtin_elementwise_max(h, h * c01);   // leaky relu (v_pk_max_num_f16)
          bv.h2[i] = h;
        }
        Hb[kb] = bv.v;
      }

      // h2^T = w2^T @ h1^T : D rows = output neurons j, D cols = events m (per lane)
      // layer 3 is then a purely per-lane dot with w3
      float acc = 0.f;
      #pragma unroll
      for (int jt = 0; jt < 7; ++jt) {
        v8f d = {0.f, 0.f, 0.f, 0.f, 0.f, 0.f, 0.f, 0.f};
        #pragma unroll
        for (int kb = 0; kb < 4; ++kb) {
          d = __builtin_amdgcn_wmma_f32_16x16x32_f16(
                false, W2A[jt * 4 + kb], false, Hb[kb], (short)0, d, false, false);
        }
        #pragma unroll
        for (int r = 0; r < 8; ++r) {
          float h = d[r];                       // already includes b2 (folded row)
          h = fmaxf(h, SLOPEF * h);             // leaky relu
          acc = fmaf(h, w3r[jt * 8 + r], acc);
        }
      }
      // combine the two 16-lane halves (j in [0..7]+jt*16 vs [8..15]+jt*16)
      acc += __shfl_xor(acc, 16);
      if (valid && lane < 16) {
        atomicAdd(out + idx0 + HH * WW * c, tn * (acc + b3v));
      }
    }
  }
}

extern "C" void kernel_launch(void* const* d_in, const int* in_sizes, int n_in,
                              void* d_out, int out_size, void* d_ws, size_t ws_size,
                              hipStream_t stream) {
  const float* t  = (const float*)d_in[0];
  const int*   x  = (const int*)d_in[1];
  const int*   y  = (const int*)d_in[2];
  const int*   p  = (const int*)d_in[3];
  const int*   b  = (const int*)d_in[4];
  // d_in[5] is the scalar B (=16), constants are hardcoded
  const float* w1 = (const float*)d_in[6];
  const float* b1 = (const float*)d_in[7];
  const float* w2 = (const float*)d_in[8];
  const float* b2 = (const float*)d_in[9];
  const float* w3 = (const float*)d_in[10];
  const float* b3 = (const float*)d_in[11];
  int n = in_sizes[0];
  unsigned char* ws = (unsigned char*)d_ws;

  (void)hipMemsetAsync(d_out, 0, (size_t)out_size * sizeof(float), stream);
  prep_kernel<<<120, 256, 0, stream>>>(w1, b1, w2, b2, w3, b3, ws);
  binmax_kernel<<<(n + 255) / 256, 256, 0, stream>>>(t, b, n, (unsigned*)(ws + OFF_TMAX));

  int ntiles = (n + 15) >> 4;
  int blocks = (ntiles + 7) / 8;
  if (blocks > 2048) blocks = 2048;
  evmlp_kernel<<<blocks, 256, 0, stream>>>(t, x, y, p, b, ws, (float*)d_out, n);
}
